// Attention_15762529976773
// MI455X (gfx1250) — matrix operand order repaired and checked by reading, not compiled, not run on hardware
//
#include <hip/hip_runtime.h>
#include <hip/hip_bf16.h>

// ---------------------------------------------------------------------------
// Problem constants (from reference): B=4, S=2048, HID=896, H=14, KV=2, D=64
// ---------------------------------------------------------------------------
#define BATCH 4
#define SEQ   2048
#define HID   896
#define NH    14
#define NKV   2
#define DH    64
#define REP   (NH / NKV)           // 7
#define MS    (BATCH * SEQ)        // 8192 rows
#define WAVES 8                    // 256-thread blocks = 8 wave32s

typedef __attribute__((ext_vector_type(16))) _Float16 v16h;
typedef __attribute__((ext_vector_type(8)))  float    v8f;

// Fragment: 8 VGPRs = 16 halfs = two 16-byte chunks
union FragH {
    v16h v;
    uint4 q[2];
};

static __device__ __forceinline__ v8f vzero8() {
    v8f z = {0.f, 0.f, 0.f, 0.f, 0.f, 0.f, 0.f, 0.f};
    return z;
}

// ---------------------------------------------------------------------------
// Elementwise converters
// ---------------------------------------------------------------------------
__global__ void athena_cvt_f16(const float* __restrict__ src,
                               _Float16* __restrict__ dst, size_t n) {
    size_t i = (size_t)blockIdx.x * blockDim.x + threadIdx.x;
    if (i < n) dst[i] = (_Float16)src[i];
}

// W is K x N (row major). Write WT as N x K (row major) in f16.
__global__ void athena_transpose_f16(const float* __restrict__ W,
                                     _Float16* __restrict__ WT,
                                     int K, int N) {
    size_t i = (size_t)blockIdx.x * blockDim.x + threadIdx.x;
    size_t total = (size_t)K * N;
    if (i >= total) return;
    int k = (int)(i / N);
    int n = (int)(i % N);
    WT[(size_t)n * K + k] = (_Float16)W[i];
}

// ---------------------------------------------------------------------------
// RoPE for Q: f32 [b,s,h,d] -> f16 same layout, rotated pairs
// ---------------------------------------------------------------------------
__global__ void athena_rope_q(const float* __restrict__ qf,
                              const float* __restrict__ cs,
                              const float* __restrict__ sn,
                              _Float16* __restrict__ qh) {
    size_t i = (size_t)blockIdx.x * blockDim.x + threadIdx.x;
    const size_t npairs = (size_t)BATCH * SEQ * NH * (DH / 2);
    if (i >= npairs) return;
    int p = (int)(i & 31);          // DH/2 = 32
    size_t t = i >> 5;
    int h = (int)(t % NH); t /= NH;
    int s = (int)(t % SEQ);
    int b = (int)(t / SEQ);
    size_t base = ((size_t)(b * SEQ + s)) * (NH * DH) + (size_t)h * DH + 2 * p;
    float t0 = qf[base], t1 = qf[base + 1];
    float c = cs[s * 32 + p], si = sn[s * 32 + p];
    qh[base]     = (_Float16)(t0 * c - t1 * si);
    qh[base + 1] = (_Float16)(t0 * si + t1 * c);
}

// RoPE for K: f32 [b,s,kv,d] -> f16 [b,kv,s,d]
__global__ void athena_rope_k(const float* __restrict__ kf,
                              const float* __restrict__ cs,
                              const float* __restrict__ sn,
                              _Float16* __restrict__ kh) {
    size_t i = (size_t)blockIdx.x * blockDim.x + threadIdx.x;
    const size_t npairs = (size_t)BATCH * SEQ * NKV * (DH / 2);
    if (i >= npairs) return;
    int p = (int)(i & 31);
    size_t t = i >> 5;
    int kv = (int)(t % NKV); t /= NKV;
    int s  = (int)(t % SEQ);
    int b  = (int)(t / SEQ);
    size_t src = ((size_t)(b * SEQ + s)) * (NKV * DH) + (size_t)kv * DH + 2 * p;
    size_t dst = (((size_t)(b * NKV + kv)) * SEQ + s) * DH + 2 * p;
    float t0 = kf[src], t1 = kf[src + 1];
    float c = cs[s * 32 + p], si = sn[s * 32 + p];
    kh[dst]     = (_Float16)(t0 * c - t1 * si);
    kh[dst + 1] = (_Float16)(t0 * si + t1 * c);
}

// V: f32 [b,s,kv,d] -> f16 transposed [b,kv,d,s]
__global__ void athena_cvt_v(const float* __restrict__ vf,
                             _Float16* __restrict__ vT) {
    size_t i = (size_t)blockIdx.x * blockDim.x + threadIdx.x;
    const size_t n = (size_t)BATCH * SEQ * NKV * DH;
    if (i >= n) return;
    int d = (int)(i & 63);
    size_t t = i >> 6;
    int kv = (int)(t % NKV); t /= NKV;
    int s  = (int)(t % SEQ);
    int b  = (int)(t / SEQ);
    size_t src = ((size_t)(b * SEQ + s)) * (NKV * DH) + (size_t)kv * DH + d;
    size_t dst = (((size_t)(b * NKV + kv)) * DH + d) * SEQ + s;
    vT[dst] = (_Float16)vf[src];
}

// ---------------------------------------------------------------------------
// Generic f16 WMMA GEMM: C[MxN] (f32) = A[MxK] (f16, row major) * B,
// where B is supplied transposed as BT[NxK] (f16, row major).
// One wave computes a 16x64 output tile (A-fragment reused 4x).
// Requires M%16==0, N%64==0, K%32==0.
// ---------------------------------------------------------------------------
__global__ __launch_bounds__(256) void athena_gemm_f16(
        const _Float16* __restrict__ A,
        const _Float16* __restrict__ BT,
        float* __restrict__ C,
        int M, int N, int K) {
    const int lane = threadIdx.x & 31;
    const int wv   = threadIdx.x >> 5;
    const int gw   = blockIdx.x * WAVES + wv;
    const int tmN  = M >> 4;
    const int tnN  = N >> 6;
    if (gw >= tmN * tnN) return;              // wave-uniform
    const int tm = gw % tmN;
    const int tn = gw / tmN;

    const int m   = lane & 15;
    const int khf = lane >> 4;                // which K-half this lane holds
    const _Float16* arow = A + (size_t)(tm * 16 + m) * K;
    const _Float16* brow0 = BT + (size_t)(tn * 64 + m) * K + khf * 16;

    v8f acc0 = vzero8(), acc1 = vzero8(), acc2 = vzero8(), acc3 = vzero8();

    for (int kk = 0; kk < K; kk += 32) {
        __builtin_prefetch(arow + kk + 128, 0, 0);   // global_prefetch_b8
        FragH a;
        a.q[0] = *(const uint4*)(arow + kk + khf * 16);      // K 0-7 / 16-23
        a.q[1] = *(const uint4*)(arow + kk + khf * 16 + 8);  // K 8-15 / 24-31
        FragH b0, b1, b2, b3;
        b0.q[0] = *(const uint4*)(brow0 + kk);
        b0.q[1] = *(const uint4*)(brow0 + kk + 8);
        b1.q[0] = *(const uint4*)(brow0 + 16 * K + kk);
        b1.q[1] = *(const uint4*)(brow0 + 16 * K + kk + 8);
        b2.q[0] = *(const uint4*)(brow0 + 32 * K + kk);
        b2.q[1] = *(const uint4*)(brow0 + 32 * K + kk + 8);
        b3.q[0] = *(const uint4*)(brow0 + 48 * K + kk);
        b3.q[1] = *(const uint4*)(brow0 + 48 * K + kk + 8);
        acc0 = __builtin_amdgcn_wmma_f32_16x16x32_f16(false, a.v, false, b0.v,
                                                      (short)0, acc0, false, false);
        acc1 = __builtin_amdgcn_wmma_f32_16x16x32_f16(false, a.v, false, b1.v,
                                                      (short)0, acc1, false, false);
        acc2 = __builtin_amdgcn_wmma_f32_16x16x32_f16(false, a.v, false, b2.v,
                                                      (short)0, acc2, false, false);
        acc3 = __builtin_amdgcn_wmma_f32_16x16x32_f16(false, a.v, false, b3.v,
                                                      (short)0, acc3, false, false);
    }

    // C layout: element r of the v8f lives at row (lane>>4)*8 + r, col lane&15
    const int mb = (lane >> 4) * 8;
    const int n0 = tn * 64 + (lane & 15);
    for (int r = 0; r < 8; ++r) {
        float* crow = C + (size_t)(tm * 16 + mb + r) * N;
        crow[n0]      = acc0[r];
        crow[n0 + 16] = acc1[r];
        crow[n0 + 32] = acc2[r];
        crow[n0 + 48] = acc3[r];
    }
}

// ---------------------------------------------------------------------------
// Flash attention: one wave per (b, h, 16-query tile).
// qh  : f16 [b,s,h,d]       (row stride NH*DH = 896)
// kh  : f16 [b,kv,s,d]      (row stride DH)
// vT  : f16 [b,kv,d,s]      (row stride SEQ)
// aout: f16 [b,s,h,d]       attention output (input to Wo GEMM)
// ---------------------------------------------------------------------------
__global__ __launch_bounds__(256) void athena_flash(
        const _Float16* __restrict__ qh,
        const _Float16* __restrict__ kh,
        const _Float16* __restrict__ vT,
        _Float16* __restrict__ aout) {
    __shared__ __align__(16) _Float16 smem[WAVES][16 * 32];   // P tile per wave

    const int lane = threadIdx.x & 31;
    const int wv   = threadIdx.x >> 5;
    const int gw   = blockIdx.x * WAVES + wv;
    const int qtN  = SEQ / 16;                 // 128
    const int total = BATCH * NH * qtN;        // 7168 waves
    if (gw >= total) return;                   // wave-uniform

    const int qt = gw % qtN;
    const int h  = (gw / qtN) % NH;
    const int b  = gw / (qtN * NH);
    const int q0 = qt * 16;
    const int kv = h / REP;

    const _Float16* qptr = qh + ((size_t)(b * SEQ + q0)) * (NH * DH) + (size_t)h * DH;
    const _Float16* kptr = kh + ((size_t)(b * NKV + kv)) * SEQ * DH;
    const _Float16* vptr = vT + ((size_t)(b * NKV + kv)) * DH * SEQ;

    const int n   = lane & 15;
    const int khf = lane >> 4;

    // Q fragments for both 32-wide D slices (loaded once)
    FragH aq0, aq1;
    {
        const _Float16* qr = qptr + (size_t)n * (NH * DH);
        aq0.q[0] = *(const uint4*)(qr + khf * 16);
        aq0.q[1] = *(const uint4*)(qr + khf * 16 + 8);
        aq1.q[0] = *(const uint4*)(qr + 32 + khf * 16);
        aq1.q[1] = *(const uint4*)(qr + 32 + khf * 16 + 8);
    }

    v8f o0 = vzero8(), o1 = vzero8(), o2 = vzero8(), o3 = vzero8();
    float mrow[8], lrow[8];
    for (int r = 0; r < 8; ++r) { mrow[r] = -1e30f; lrow[r] = 0.f; }

    const int   mb    = khf * 8;            // C-frag row base for this lane
    const float scale = 0.125f;             // 1/sqrt(64)
    _Float16*   ms    = &smem[wv][0];
    const int   kend  = q0 + 16;            // causal horizon (exclusive)

    for (int key0 = 0; key0 < kend; key0 += 32) {
        // ---- scores: S = Q K^T, two 16-key sub-tiles -----------------------
        v8f s0 = vzero8(), s1 = vzero8();
        {
            const _Float16* kr0 = kptr + (size_t)(key0 + n) * DH;
            const _Float16* kr1 = kptr + (size_t)(key0 + 16 + n) * DH;
            FragH bk;
            bk.q[0] = *(const uint4*)(kr0 + khf * 16);
            bk.q[1] = *(const uint4*)(kr0 + khf * 16 + 8);
            s0 = __builtin_amdgcn_wmma_f32_16x16x32_f16(false, aq0.v, false, bk.v,
                                                        (short)0, s0, false, false);
            bk.q[0] = *(const uint4*)(kr0 + 32 + khf * 16);
            bk.q[1] = *(const uint4*)(kr0 + 32 + khf * 16 + 8);
            s0 = __builtin_amdgcn_wmma_f32_16x16x32_f16(false, aq1.v, false, bk.v,
                                                        (short)0, s0, false, false);
            bk.q[0] = *(const uint4*)(kr1 + khf * 16);
            bk.q[1] = *(const uint4*)(kr1 + khf * 16 + 8);
            s1 = __builtin_amdgcn_wmma_f32_16x16x32_f16(false, aq0.v, false, bk.v,
                                                        (short)0, s1, false, false);
            bk.q[0] = *(const uint4*)(kr1 + 32 + khf * 16);
            bk.q[1] = *(const uint4*)(kr1 + 32 + khf * 16 + 8);
            s1 = __builtin_amdgcn_wmma_f32_16x16x32_f16(false, aq1.v, false, bk.v,
                                                        (short)0, s1, false, false);
        }

        // ---- online softmax (causal mask applied analytically) -------------
        for (int r = 0; r < 8; ++r) {
            const int qidx = q0 + mb + r;
            float a0 = s0[r] * scale + ((key0 + n)      <= qidx ? 0.f : -1e9f);
            float a1 = s1[r] * scale + ((key0 + 16 + n) <= qidx ? 0.f : -1e9f);

            float mx = fmaxf(a0, a1);
            mx = fmaxf(mx, __shfl_xor(mx, 1, 32));
            mx = fmaxf(mx, __shfl_xor(mx, 2, 32));
            mx = fmaxf(mx, __shfl_xor(mx, 4, 32));
            mx = fmaxf(mx, __shfl_xor(mx, 8, 32));

            float mnew  = fmaxf(mrow[r], mx);
            float alpha = __expf(mrow[r] - mnew);
            mrow[r] = mnew;

            float p0 = __expf(a0 - mnew);
            float p1 = __expf(a1 - mnew);
            float rs = p0 + p1;
            rs += __shfl_xor(rs, 1, 32);
            rs += __shfl_xor(rs, 2, 32);
            rs += __shfl_xor(rs, 4, 32);
            rs += __shfl_xor(rs, 8, 32);
            lrow[r] = lrow[r] * alpha + rs;

            o0[r] *= alpha; o1[r] *= alpha; o2[r] *= alpha; o3[r] *= alpha;

            // stash P tile (16 queries x 32 keys) in wave-private LDS
            ms[(mb + r) * 32 + n]      = (_Float16)p0;
            ms[(mb + r) * 32 + 16 + n] = (_Float16)p1;
        }

        // ---- re-layout P via LDS into an A fragment ------------------------
        FragH ap;
        {
            const uint4* lp = (const uint4*)(ms + n * 32);
            ap.q[0] = lp[2 * khf];       // K 0-7 / 16-23 of the 32-key block
            ap.q[1] = lp[2 * khf + 1];   // K 8-15 / 24-31
        }

        // ---- O += P * V (K-dim = 32 keys), 4 D-slices ----------------------
        {
            const _Float16* vr = vptr + (size_t)n * SEQ + key0 + khf * 16;
            FragH bv;
            bv.q[0] = *(const uint4*)(vr);
            bv.q[1] = *(const uint4*)(vr + 8);
            o0 = __builtin_amdgcn_wmma_f32_16x16x32_f16(false, ap.v, false, bv.v,
                                                        (short)0, o0, false, false);
            bv.q[0] = *(const uint4*)(vr + 16 * SEQ);
            bv.q[1] = *(const uint4*)(vr + 16 * SEQ + 8);
            o1 = __builtin_amdgcn_wmma_f32_16x16x32_f16(false, ap.v, false, bv.v,
                                                        (short)0, o1, false, false);
            bv.q[0] = *(const uint4*)(vr + 32 * SEQ);
            bv.q[1] = *(const uint4*)(vr + 32 * SEQ + 8);
            o2 = __builtin_amdgcn_wmma_f32_16x16x32_f16(false, ap.v, false, bv.v,
                                                        (short)0, o2, false, false);
            bv.q[0] = *(const uint4*)(vr + 48 * SEQ);
            bv.q[1] = *(const uint4*)(vr + 48 * SEQ + 8);
            o3 = __builtin_amdgcn_wmma_f32_16x16x32_f16(false, ap.v, false, bv.v,
                                                        (short)0, o3, false, false);
        }
    }

    // ---- normalize and store (f16, [b,s,h,d] layout for the Wo GEMM) -------
    for (int r = 0; r < 8; ++r) {
        float inv = 1.0f / lrow[r];
        _Float16* orow = aout + ((size_t)(b * SEQ + q0 + mb + r)) * (NH * DH)
                              + (size_t)h * DH + n;
        orow[0]  = (_Float16)(o0[r] * inv);
        orow[16] = (_Float16)(o1[r] * inv);
        orow[32] = (_Float16)(o2[r] * inv);
        orow[48] = (_Float16)(o3[r] * inv);
    }
}

// ---------------------------------------------------------------------------
// Host side
// ---------------------------------------------------------------------------
extern "C" void kernel_launch(void* const* d_in, const int* in_sizes, int n_in,
                              void* d_out, int out_size, void* d_ws, size_t ws_size,
                              hipStream_t stream) {
    (void)in_sizes; (void)n_in; (void)out_size; (void)ws_size;

    const float* x  = (const float*)d_in[0];
    const float* fc = (const float*)d_in[1];
    const float* fs = (const float*)d_in[2];
    /* d_in[3] = mask: causal, applied analytically in-kernel */
    const float* Wq = (const float*)d_in[4];
    const float* Wk = (const float*)d_in[5];
    const float* Wv = (const float*)d_in[6];
    const float* Wo = (const float*)d_in[7];
    float* out = (float*)d_out;

    char* ws = (char*)d_ws;
    size_t off = 0;
    auto take = [&](size_t bytes) -> char* {
        char* p = ws + off;
        off = (off + bytes + 255) & ~(size_t)255;
        return p;
    };

    _Float16* xh   = (_Float16*)take((size_t)MS * HID * sizeof(_Float16));
    _Float16* WqT  = (_Float16*)take((size_t)(NH * DH) * HID * sizeof(_Float16));
    _Float16* WkT  = (_Float16*)take((size_t)(NKV * DH) * HID * sizeof(_Float16));
    _Float16* WvT  = (_Float16*)take((size_t)(NKV * DH) * HID * sizeof(_Float16));
    _Float16* WoT  = (_Float16*)take((size_t)HID * (NH * DH) * sizeof(_Float16));
    float*    qf   = (float*)   take((size_t)MS * (NH * DH) * sizeof(float));
    float*    kf   = (float*)   take((size_t)MS * (NKV * DH) * sizeof(float));
    float*    vf   = (float*)   take((size_t)MS * (NKV * DH) * sizeof(float));
    _Float16* qhb  = (_Float16*)take((size_t)MS * (NH * DH) * sizeof(_Float16));
    _Float16* khb  = (_Float16*)take((size_t)MS * (NKV * DH) * sizeof(_Float16));
    _Float16* vTb  = (_Float16*)take((size_t)MS * (NKV * DH) * sizeof(_Float16));
    _Float16* aout = xh;   // xh dead after the QKV GEMMs -> reuse for attn out

    const int T = 256;
    auto b1d = [](size_t nel, int t) { return (unsigned)((nel + t - 1) / t); };

    // 1) f32 -> f16 conversions (weights transposed for the B-fragment path)
    athena_cvt_f16<<<b1d((size_t)MS * HID, T), T, 0, stream>>>(x, xh, (size_t)MS * HID);
    athena_transpose_f16<<<b1d((size_t)HID * NH * DH, T), T, 0, stream>>>(Wq, WqT, HID, NH * DH);
    athena_transpose_f16<<<b1d((size_t)HID * NKV * DH, T), T, 0, stream>>>(Wk, WkT, HID, NKV * DH);
    athena_transpose_f16<<<b1d((size_t)HID * NKV * DH, T), T, 0, stream>>>(Wv, WvT, HID, NKV * DH);
    athena_transpose_f16<<<b1d((size_t)(NH * DH) * HID, T), T, 0, stream>>>(Wo, WoT, NH * DH, HID);

    // 2) QKV projections (WMMA GEMMs)
    {
        int tiles_q = (MS / 16) * ((NH * DH) / 64);     // 512*14 = 7168
        int tiles_k = (MS / 16) * ((NKV * DH) / 64);    // 512*2  = 1024
        athena_gemm_f16<<<(tiles_q + WAVES - 1) / WAVES, T, 0, stream>>>(
            xh, WqT, qf, MS, NH * DH, HID);
        athena_gemm_f16<<<(tiles_k + WAVES - 1) / WAVES, T, 0, stream>>>(
            xh, WkT, kf, MS, NKV * DH, HID);
        athena_gemm_f16<<<(tiles_k + WAVES - 1) / WAVES, T, 0, stream>>>(
            xh, WvT, vf, MS, NKV * DH, HID);
    }

    // 3) RoPE + layout conversion to f16 operands
    athena_rope_q<<<b1d((size_t)BATCH * SEQ * NH * 32, T), T, 0, stream>>>(qf, fc, fs, qhb);
    athena_rope_k<<<b1d((size_t)BATCH * SEQ * NKV * 32, T), T, 0, stream>>>(kf, fc, fs, khb);
    athena_cvt_v <<<b1d((size_t)BATCH * SEQ * NKV * DH, T), T, 0, stream>>>(vf, vTb);

    // 4) Flash attention (WMMA): one wave per (b, h, 16-query tile)
    {
        int waves = BATCH * NH * (SEQ / 16);            // 7168
        athena_flash<<<(waves + WAVES - 1) / WAVES, T, 0, stream>>>(qhb, khb, vTb, aout);
    }

    // 5) Output projection -> f32 d_out
    {
        int tiles = (MS / 16) * (HID / 64);
        athena_gemm_f16<<<(tiles + WAVES - 1) / WAVES, T, 0, stream>>>(
            aout, WoT, out, MS, HID, HID);
    }
}
